// LoRAAtoms_2327872275247
// MI455X (gfx1250) — compile-verified, matrix operand order checked by reading, not run on hardware
//
#include <hip/hip_runtime.h>
#include <hip/hip_bf16.h>

// ---------------------------------------------------------------------------
// LoRA atoms:  out[b,s,p] = sum_a scale[b,a] * (hidden[b,s,:] @ A[a,l,:,:]) @ B[a,l,:,:]
// Fused single-kernel formulation over the joined axis k = a*6 + r (384):
//   Stage 1: P[16, 384]   = hidden_tile(16 x 2048) @ A2(2048 x 384)   (K = 2048)
//            scaled by scale[b, k/6], deposited as bf16 in LDS.
//   Stage 2: out[16, 2048] = P(16 x 384) @ B2(384 x 2048)             (K = 384)
// Both stages use V_WMMA_F32_16X16X32_BF16 with f32 accumulation. Fusion keeps
// P entirely in LDS (12 KB) -> no 25 MB HBM round trip for the intermediate.
// HBM traffic ~ hidden 128MB + out 128MB + packed weights; floor ~11 us @ 23.3 TB/s.
// bf16 WMMA (K=32/instr) keeps compute far below that floor; the f32 WMMA path
// (16x16x4) would risk becoming the bottleneck.
// ---------------------------------------------------------------------------

typedef __attribute__((ext_vector_type(16))) __bf16 v16bf;
typedef __attribute__((ext_vector_type(8)))  float  v8f;

union Frag16 { unsigned int u[8]; v16bf v; };

__device__ __forceinline__ unsigned short f32_bf16(float f) {
    unsigned int u = __builtin_bit_cast(unsigned int, f);
    u += 0x7FFFu + ((u >> 16) & 1u);          // round-to-nearest-even
    return (unsigned short)(u >> 16);
}

// ISA 16-bit A/B fragment layout (CDNA5 ISA §7.12.2):
// lane L holds row/col (L&15); VGPR j holds the bf16 pair starting at
//   K0 = (j>=4 ? 16 : 0) + ((L>>4)*8) + (j&3)*2
__device__ __forceinline__ int kpair_base(int lane, int j) {
    return ((j >> 2) << 4) + ((lane >> 4) << 3) + ((j & 3) << 1);
}

// ---------------------------------------------------------------------------
// Pack kernel: bf16, K-pair-contiguous weight panels in workspace.
//   Abf[k][d]  (k=384, d=2048):  Abf[k*2048+d] = A[a, layer, d, r]
//   Bbf[p][k]  (p=2048, k=384):  Bbf[p*384 +k] = B[a, layer, r, p]
// layer_idx lives on device (graph-capture safe) -> read inside the kernel.
// ---------------------------------------------------------------------------
__global__ void __launch_bounds__(256)
lora_pack(const float* __restrict__ A, const float* __restrict__ Bm,
          const int* __restrict__ layer_p,
          unsigned short* __restrict__ Abf, unsigned short* __restrict__ Bbf) {
    const int layer = *layer_p;
    const int idx = blockIdx.x * 256 + threadIdx.x;   // 384*2048 threads total
    const int k = idx >> 11;                          // 0..383
    const int d = idx & 2047;                         // 0..2047 (doubles as 'p')
    const int a = k / 6;
    const int r = k - a * 6;
    // A[a, layer, d, r]  strides: a:(16*2048*6) layer:(2048*6) d:6 r:1
    Abf[(size_t)k * 2048 + d] =
        f32_bf16(A[((size_t)(a * 16 + layer) * 2048 + d) * 6 + r]);
    // B[a, layer, r, p]  strides: a:(16*6*2048) layer:(6*2048) r:2048 p:1
    Bbf[(size_t)d * 384 + k] =
        f32_bf16(Bm[((size_t)(a * 16 + layer) * 6 + r) * 2048 + d]);
}

// ---------------------------------------------------------------------------
// Fused kernel: one block = one (b, 16-row seq tile); 256 threads = 8 waves.
// Stage 1: wave w owns P columns [w*48, w*48+48) (3 WMMA tiles). Hidden is
//          staged 16x64 f32 -> bf16 in LDS per barrier (6 WMMAs / barrier).
// Stage 2: wave w owns out columns [w*256, w*256+256) (16 tiles, 128 acc VGPRs).
// ---------------------------------------------------------------------------
__global__ void __launch_bounds__(256)
lora_fused(const float* __restrict__ hidden, const float* __restrict__ scales,
           const unsigned short* __restrict__ Abf, const unsigned short* __restrict__ Bbf,
           float* __restrict__ out) {
    __shared__ unsigned int ldsH[16 * 32];                       // 16 rows x 64 bf16 (32 pairs)
    __shared__ __align__(16) unsigned short ldsP[16 * 384];      // P tile, bf16

    const int b     = blockIdx.y;
    const int s0    = blockIdx.x * 16;
    const int tid   = threadIdx.x;
    const int lane  = tid & 31;
    const int wave  = tid >> 5;
    const int lrow  = lane & 15;
    const int mbase = (lane >> 4) << 3;   // C/D layout: VGPR j -> M = mbase + j

    const v8f vzero = {0.f,0.f,0.f,0.f,0.f,0.f,0.f,0.f};

    // ------------------------- Stage 1: K = 2048 --------------------------
    {
        const int col0 = wave * 48 + lrow;
        v8f acc0 = vzero, acc1 = vzero, acc2 = vzero;

        // Staging role: thread handles row = tid>>4, pairs pr and pr+16.
        const int srow = tid >> 4;
        const int spr  = tid & 15;
        const float* hrow = hidden + ((size_t)b * 4096 + s0 + srow) * 2048;

        for (int d0 = 0; d0 < 2048; d0 += 64) {
            __syncthreads();                      // protect LDS reuse
            {
                const float* hp = hrow + d0;
                if (d0 + 64 < 2048) __builtin_prefetch(hp + 64, 0, 0);  // global_prefetch_b8
                unsigned int l0 = f32_bf16(hp[spr * 2]);
                unsigned int h0 = f32_bf16(hp[spr * 2 + 1]);
                unsigned int l1 = f32_bf16(hp[32 + spr * 2]);
                unsigned int h1 = f32_bf16(hp[32 + spr * 2 + 1]);
                ldsH[srow * 32 + spr]      = l0 | (h0 << 16);
                ldsH[srow * 32 + spr + 16] = l1 | (h1 << 16);
            }
            __syncthreads();

            #pragma unroll
            for (int half = 0; half < 2; ++half) {
                const int kofs = half * 32;
                Frag16 a;
                #pragma unroll
                for (int j = 0; j < 8; ++j)
                    a.u[j] = ldsH[lrow * 32 + ((kofs + kpair_base(lane, j)) >> 1)];

                Frag16 b0, b1, b2;
                #pragma unroll
                for (int j = 0; j < 8; ++j) {
                    const int kk = d0 + kofs + kpair_base(lane, j);
                    b0.u[j] = *(const unsigned int*)(Abf + (size_t)(col0     ) * 2048 + kk);
                    b1.u[j] = *(const unsigned int*)(Abf + (size_t)(col0 + 16) * 2048 + kk);
                    b2.u[j] = *(const unsigned int*)(Abf + (size_t)(col0 + 32) * 2048 + kk);
                }
                acc0 = __builtin_amdgcn_wmma_f32_16x16x32_bf16(false, a.v, false, b0.v, (short)0, acc0, false, false);
                acc1 = __builtin_amdgcn_wmma_f32_16x16x32_bf16(false, a.v, false, b1.v, (short)0, acc1, false, false);
                acc2 = __builtin_amdgcn_wmma_f32_16x16x32_bf16(false, a.v, false, b2.v, (short)0, acc2, false, false);
            }
        }

        // Scale by atom_scales[b, col/6] and deposit bf16 P tile into LDS.
        #pragma unroll
        for (int t = 0; t < 3; ++t) {
            const int col = col0 + t * 16;
            const float sc = scales[b * 64 + col / 6];
            const v8f c = (t == 0) ? acc0 : ((t == 1) ? acc1 : acc2);
            #pragma unroll
            for (int j = 0; j < 8; ++j)
                ldsP[(mbase + j) * 384 + col] = f32_bf16(c[j] * sc);
        }
    }
    __syncthreads();   // P tile complete before stage 2 reads it

    // ------------------------- Stage 2: K = 384 ---------------------------
    {
        const unsigned int* ldsPu = (const unsigned int*)ldsP;   // 192 pairs/row
        const int colbase = wave * 256 + lrow;

        v8f acc[16];
        #pragma unroll
        for (int t = 0; t < 16; ++t) acc[t] = vzero;

        for (int k0 = 0; k0 < 384; k0 += 32) {
            Frag16 a;
            #pragma unroll
            for (int j = 0; j < 8; ++j)
                a.u[j] = ldsPu[lrow * 192 + ((k0 + kpair_base(lane, j)) >> 1)];

            #pragma unroll
            for (int t = 0; t < 16; ++t) {
                const int col = colbase + t * 16;
                const unsigned short* bp = Bbf + (size_t)col * 384 + k0;
                Frag16 bb;
                #pragma unroll
                for (int j = 0; j < 8; ++j)
                    bb.u[j] = *(const unsigned int*)(bp + kpair_base(lane, j));
                acc[t] = __builtin_amdgcn_wmma_f32_16x16x32_bf16(false, a.v, false, bb.v, (short)0, acc[t], false, false);
            }
        }

        #pragma unroll
        for (int t = 0; t < 16; ++t) {
            const int col = colbase + t * 16;
            #pragma unroll
            for (int j = 0; j < 8; ++j)
                out[((size_t)b * 4096 + s0 + mbase + j) * 2048 + col] = acc[t][j];
        }
    }
}

// ---------------------------------------------------------------------------
// Launch: pack -> fused GEMM, both on `stream` (same-stream ordering).
// Workspace layout (bytes):
//   [0      , 1572864)  Abf : 384*2048 bf16   (A2 panel, K-pair contiguous)
//   [1572864, 3145728)  Bbf : 2048*384 bf16   (B2 panel, K-pair contiguous)
// ---------------------------------------------------------------------------
extern "C" void kernel_launch(void* const* d_in, const int* in_sizes, int n_in,
                              void* d_out, int out_size, void* d_ws, size_t ws_size,
                              hipStream_t stream) {
    const float* hidden = (const float*)d_in[0];   // (4,4096,2048)
    const float* scales = (const float*)d_in[1];   // (4,64)
    const float* A      = (const float*)d_in[2];   // (64,16,2048,6)
    const float* Bm     = (const float*)d_in[3];   // (64,16,6,2048)
    const int*   layer  = (const int*)d_in[4];     // scalar on device

    unsigned short* Abf = (unsigned short*)d_ws;
    unsigned short* Bbf = (unsigned short*)((char*)d_ws + 1572864);
    float* out = (float*)d_out;

    lora_pack <<<dim3(3072),   dim3(256), 0, stream>>>(A, Bm, layer, Abf, Bbf);
    lora_fused<<<dim3(256, 4), dim3(256), 0, stream>>>(hidden, scales, Abf, Bbf, out);
}
